// Mamba2_Norm_29222957482775
// MI455X (gfx1250) — matrix operand order repaired and checked by reading, not compiled, not run on hardware
//
#include <hip/hip_runtime.h>
#include <hip/hip_bf16.h>
#include <stdint.h>

// ---------------------------------------------------------------- constants
#define DMODEL   2048
#define DINNER   4096
#define HEADDIM_ 64
#define NHEADS_  64
#define DSTATE   128
#define DCONV    4
#define CHUNK_   256
#define NCHUNK_  8
#define BATCH_   2
#define SEQLEN_  2048
#define BL_      (BATCH_ * SEQLEN_)                   // 4096
#define CONVDIM  (DINNER + 2 * DSTATE)                // 4352
#define DPROJ    (2 * DINNER + 2 * DSTATE + NHEADS_)  // 8512

typedef __bf16 bf16;
typedef __attribute__((ext_vector_type(16))) __bf16 v16bf;
typedef __attribute__((ext_vector_type(8)))  float  v8f;
typedef int v4i_g __attribute__((vector_size(16)));   // matches builtin's pointee type

struct Frag { uint32_t u[8]; };

// CDNA5 async global->LDS path (ASYNCcnt) if the toolchain exposes it.
#if __has_builtin(__builtin_amdgcn_global_load_async_to_lds_b128) && \
    __has_builtin(__builtin_amdgcn_s_wait_asynccnt)
#define HAVE_ASYNC_LDS 1
#define TO_LDS(p) ((__attribute__((address_space(3))) v4i_g*)(void*)(p))
#define TO_GLB(p) ((__attribute__((address_space(1))) v4i_g*)(void*)(p))
#else
#define HAVE_ASYNC_LDS 0
#endif

// ---------------------------------------------------------------- helpers
__device__ __forceinline__ float bf2f(bf16 x) {
  uint16_t h = __builtin_bit_cast(uint16_t, x);
  uint32_t u = ((uint32_t)h) << 16;
  return __builtin_bit_cast(float, u);
}
__device__ __forceinline__ bf16 f2bf(float f) {
  uint32_t u = __builtin_bit_cast(uint32_t, f);
  u += 0x7FFFu + ((u >> 16) & 1u);                    // round-to-nearest-even
  uint16_t h = (uint16_t)(u >> 16);
  return __builtin_bit_cast(bf16, h);
}
__device__ __forceinline__ v8f wmma_bf16(v16bf a, v16bf b, v8f c) {
  return __builtin_amdgcn_wmma_f32_16x16x32_bf16(false, a, false, b, (short)0, c,
                                                 false, false);
}

// A-fragment (16x32 bf16, M x K): lane&15 = M row; lanes>=16 take K+8 half.
// Per ISA layout: elements 0..7 -> K = koff+0..7 ; elements 8..15 -> K = koff+16..23.
__device__ __forceinline__ v16bf load_a_frag(const bf16* base, int ld) {
  int lane = threadIdx.x & 31;
  const bf16* p = base + (lane & 15) * ld + ((lane >> 4) << 3);
  Frag f;
  const uint32_t* q0 = (const uint32_t*)p;
  const uint32_t* q1 = (const uint32_t*)(p + 16);
#pragma unroll
  for (int i = 0; i < 4; ++i) { f.u[i] = q0[i]; f.u[4 + i] = q1[i]; }
  return __builtin_bit_cast(v16bf, f);
}
// B-fragment (32x16 bf16, K x N) from an N-major [n][k] buffer: lane&15 = column,
// elements 0..7 -> K = (lane>>4)*8+0..7 ; elements 8..15 -> K = 16+(lane>>4)*8+0..7.
__device__ __forceinline__ v16bf load_b_frag(const bf16* base, int ld) {
  int lane = threadIdx.x & 31;
  const bf16* p = base + (lane & 15) * ld + ((lane >> 4) << 3);
  Frag f;
  const uint32_t* q0 = (const uint32_t*)p;
  const uint32_t* q1 = (const uint32_t*)(p + 16);
#pragma unroll
  for (int i = 0; i < 4; ++i) { f.u[i] = q0[i]; f.u[4 + i] = q1[i]; }
  return __builtin_bit_cast(v16bf, f);
}
// A-fragment with per-row scale (C * exp(Acum[q])).
__device__ __forceinline__ v16bf make_scaled_afrag(const bf16* row, int k0, float scale) {
  int lane = threadIdx.x & 31;
  int koff = k0 + ((lane >> 4) << 3);
  Frag f;
#pragma unroll
  for (int v = 0; v < 8; ++v) {
    int s0 = koff + ((v < 4) ? (2 * v) : (2 * v + 8));
    float g0 = bf2f(row[s0]) * scale;
    float g1 = bf2f(row[s0 + 1]) * scale;
    uint32_t lo = (uint32_t)__builtin_bit_cast(uint16_t, f2bf(g0));
    uint32_t hi = (uint32_t)__builtin_bit_cast(uint16_t, f2bf(g1));
    f.u[v] = lo | (hi << 16);
  }
  return __builtin_bit_cast(v16bf, f);
}
// A-fragment for Y_diag: G[q,s] * exp(Acum[q]-Acum[s]) masked to s<=q.
__device__ __forceinline__ v16bf make_gfrag(const bf16* row, int q, int k0, float acq,
                                            const float* sAcum) {
  int lane = threadIdx.x & 31;
  int koff = k0 + ((lane >> 4) << 3);
  Frag f;
#pragma unroll
  for (int v = 0; v < 8; ++v) {
    int s0 = koff + ((v < 4) ? (2 * v) : (2 * v + 8));
    float g0 = (s0 <= q)     ? bf2f(row[s0])     * __expf(acq - sAcum[s0])     : 0.f;
    float g1 = (s0 + 1 <= q) ? bf2f(row[s0 + 1]) * __expf(acq - sAcum[s0 + 1]) : 0.f;
    uint32_t lo = (uint32_t)__builtin_bit_cast(uint16_t, f2bf(g0));
    uint32_t hi = (uint32_t)__builtin_bit_cast(uint16_t, f2bf(g1));
    f.u[v] = lo | (hi << 16);
  }
  return __builtin_bit_cast(v16bf, f);
}

// ---------------------------------------------------------------- f32 -> bf16
__global__ __launch_bounds__(256) void f32_to_bf16_kernel(const float* __restrict__ in,
                                                          bf16* __restrict__ out, long n) {
  long i = (long)blockIdx.x * 256 + threadIdx.x;
  if (i < n) out[i] = f2bf(in[i]);
}

// ---------------------------------------------------------------- generic NT GEMM
// C[M,N] = A[M,K] @ B[N,K]^T ; bf16 in, f32 accumulate via v_wmma_f32_16x16x32_bf16.
// Block tile 128x128, BK=32, 256 threads = 8 waves in a 4(m) x 2(n) grid; each wave
// computes a 32x64 region (2x4 WMMA tiles -> 8 WMMAs per K step between barriers).
// LDS row stride 40 (pad 8) keeps 16B/32B fragment reads conflict-free.  Tiles are
// staged via GLOBAL_LOAD_ASYNC_TO_LDS_B128 (ASYNCcnt) when available.
#define GBM 128
#define GBN 128
#define GBK 32
#define LDT 40

__global__ __launch_bounds__(256) void mamba_gemm_nt(
    const bf16* __restrict__ A, const bf16* __restrict__ B,
    int M, int N, int K,
    long strideAb, long strideBb, long strideOb, int mode,
    float* __restrict__ out_f32, bf16* __restrict__ out_bf16,
    bf16* __restrict__ z_out, float* __restrict__ xbc_out, float* __restrict__ dt_out) {
  (void)M;
  __shared__ bf16 sA[GBM * LDT];
  __shared__ bf16 sB[GBN * LDT];
  const int bz = blockIdx.z;
  A += (long)bz * strideAb;
  B += (long)bz * strideBb;
  const int m0 = blockIdx.x * GBM;
  const int n0 = blockIdx.y * GBN;
  const int tid = threadIdx.x;
  const int wave = tid >> 5;
  const int lane = tid & 31;
  const int wm = wave >> 1;   // 0..3  (32 rows each)
  const int wn = wave & 1;    // 0..1  (64 cols each)
  const v8f vzero = {0.f, 0.f, 0.f, 0.f, 0.f, 0.f, 0.f, 0.f};
  v8f acc[2][4];
#pragma unroll
  for (int i = 0; i < 2; ++i)
#pragma unroll
    for (int j = 0; j < 4; ++j) acc[i][j] = vzero;

  // staging map: thread -> (row, 32B half); same for A (128 rows) and B (128 rows).
  const int srow = tid >> 1, shalf = tid & 1;
  int nr = n0 + srow;                 // clamp ragged N edge (in_proj: 8512)
  if (nr > N - 1) nr = N - 1;
  const bf16* gA = A + (long)(m0 + srow) * K + shalf * 16;
  const bf16* gB = B + (long)nr * K + shalf * 16;
  bf16* lA = sA + srow * LDT + shalf * 16;
  bf16* lB = sB + srow * LDT + shalf * 16;

  for (int k0 = 0; k0 < K; k0 += GBK) {
#if HAVE_ASYNC_LDS
    __builtin_amdgcn_global_load_async_to_lds_b128(TO_GLB(gA + k0),     TO_LDS(lA),     0, 0);
    __builtin_amdgcn_global_load_async_to_lds_b128(TO_GLB(gA + k0 + 8), TO_LDS(lA + 8), 0, 0);
    __builtin_amdgcn_global_load_async_to_lds_b128(TO_GLB(gB + k0),     TO_LDS(lB),     0, 0);
    __builtin_amdgcn_global_load_async_to_lds_b128(TO_GLB(gB + k0 + 8), TO_LDS(lB + 8), 0, 0);
    if (k0 + GBK < K) {
      __builtin_prefetch(gA + k0 + GBK, 0, 1);
      __builtin_prefetch(gB + k0 + GBK, 0, 1);
    }
    __builtin_amdgcn_s_wait_asynccnt(0);
#else
    {
      const uint32_t* src = (const uint32_t*)(gA + k0);
      uint32_t* dst = (uint32_t*)lA;
#pragma unroll
      for (int i = 0; i < 8; ++i) dst[i] = src[i];
    }
    {
      const uint32_t* src = (const uint32_t*)(gB + k0);
      uint32_t* dst = (uint32_t*)lB;
#pragma unroll
      for (int i = 0; i < 8; ++i) dst[i] = src[i];
    }
    if (k0 + GBK < K) {
      __builtin_prefetch(gA + k0 + GBK, 0, 1);
      __builtin_prefetch(gB + k0 + GBK, 0, 1);
    }
#endif
    __syncthreads();
    v16bf af[2], bfr[4];
#pragma unroll
    for (int i = 0; i < 2; ++i) af[i] = load_a_frag(sA + (wm * 32 + i * 16) * LDT, LDT);
#pragma unroll
    for (int j = 0; j < 4; ++j) bfr[j] = load_b_frag(sB + (wn * 64 + j * 16) * LDT, LDT);
#pragma unroll
    for (int i = 0; i < 2; ++i)
#pragma unroll
      for (int j = 0; j < 4; ++j) acc[i][j] = wmma_bf16(af[i], bfr[j], acc[i][j]);
    __syncthreads();
  }

#pragma unroll
  for (int i = 0; i < 2; ++i)
#pragma unroll
    for (int j = 0; j < 4; ++j)
#pragma unroll
      for (int r = 0; r < 8; ++r) {
        int row = m0 + wm * 32 + i * 16 + ((lane >> 4) << 3) + r;
        int col = n0 + wn * 64 + j * 16 + (lane & 15);
        if (col >= N) continue;
        float v = acc[i][j][r];
        if (mode == 0) {
          out_f32[(long)bz * strideOb + (long)row * N + col] = v;
        } else if (mode == 2) {
          out_bf16[(long)bz * strideOb + (long)row * N + col] = f2bf(v);
        } else {  // in_proj split epilogue: z | xBC | dt
          if (col < DINNER)
            z_out[(long)row * DINNER + col] = f2bf(v);
          else if (col < DINNER + CONVDIM)
            xbc_out[(long)row * CONVDIM + (col - DINNER)] = v;
          else
            dt_out[(long)row * NHEADS_ + (col - DINNER - CONVDIM)] = v;
        }
      }
}

// ---------------------------------------------------------------- dt softplus
__global__ __launch_bounds__(256) void dtv_kernel(const float* __restrict__ dt,
                                                  const float* __restrict__ dt_bias,
                                                  float* __restrict__ dtv) {
  int i = blockIdx.x * 256 + threadIdx.x;
  if (i >= BL_ * NHEADS_) return;
  int h = i & (NHEADS_ - 1);
  float x = dt[i] + dt_bias[h];
  dtv[i] = (x > 20.f) ? x : log1pf(__expf(x));
}

// ---------------------------------------------------------------- per-chunk cumsum of A*dt
// One block per (b,c,h); 256-thread Hillis-Steele inclusive scan.
__global__ __launch_bounds__(CHUNK_) void acum_kernel(const float* __restrict__ dtv,
                                                      const float* __restrict__ A_log,
                                                      float* __restrict__ Acum) {
  const int blk = blockIdx.x;
  const int h = blk & (NHEADS_ - 1);
  const int c = (blk >> 6) & (NCHUNK_ - 1);
  const int b = blk >> 9;
  const int s = threadIdx.x;
  const float a = -__expf(A_log[h]);
  const int m = b * SEQLEN_ + c * CHUNK_ + s;
  __shared__ float buf[CHUNK_];
  buf[s] = dtv[(long)m * NHEADS_ + h] * a;
  __syncthreads();
  for (int off = 1; off < CHUNK_; off <<= 1) {
    float t = (s >= off) ? buf[s - off] : 0.f;
    __syncthreads();
    buf[s] += t;
    __syncthreads();
  }
  Acum[((long)(b * NHEADS_ + h) * NCHUNK_ + c) * CHUNK_ + s] = buf[s];
}

// ---------------------------------------------------------------- conv1d + silu + split
__global__ __launch_bounds__(256) void convprep_kernel(
    const float* __restrict__ xBC, const float* __restrict__ conv_w,
    const float* __restrict__ conv_b, const float* __restrict__ dtv,
    bf16* __restrict__ xc, bf16* __restrict__ Bm, bf16* __restrict__ Cm) {
  long i = (long)blockIdx.x * 256 + threadIdx.x;
  if (i >= (long)BL_ * CONVDIM) return;
  int m = (int)(i / CONVDIM);
  int cch = (int)(i % CONVDIM);
  int b = m >> 11, l = m & (SEQLEN_ - 1);
  float acc = conv_b[cch];
#pragma unroll
  for (int k = 0; k < DCONV; ++k) {
    int ls = l - (DCONV - 1) + k;
    if (ls >= 0) acc += xBC[(long)(b * SEQLEN_ + ls) * CONVDIM + cch] * conv_w[cch * DCONV + k];
  }
  float sil = acc / (1.f + __expf(-acc));
  if (cch < DINNER) {
    int h = cch >> 6;
    xc[(long)m * DINNER + cch] = f2bf(sil * dtv[(long)m * NHEADS_ + h]);
  } else if (cch < DINNER + DSTATE) {
    Bm[(long)m * DSTATE + (cch - DINNER)] = f2bf(sil);
  } else {
    Cm[(long)m * DSTATE + (cch - DINNER - DSTATE)] = f2bf(sil);
  }
}

// ---------------------------------------------------------------- Y_diag per (b,c,h)
// Y = (G o tril(exp(Acum_q - Acum_s))) @ xc_h ; 256x64 output per block, 8 waves of
// 32 rows; masked K blocks are skipped wave-uniformly (EXEC stays all-ones for WMMA).
#define LDX 264  // 256 + 8 pad, stride of [p][s] LDS transpose

__global__ __launch_bounds__(256) void ydiag_kernel(const bf16* __restrict__ G,
                                                    const bf16* __restrict__ xc,
                                                    const float* __restrict__ Acum,
                                                    float* __restrict__ y_acc) {
  const int blk = blockIdx.x;
  const int h = blk & (NHEADS_ - 1);
  const int c = (blk >> 6) & (NCHUNK_ - 1);
  const int b = blk >> 9;
  const int mbase = b * SEQLEN_ + c * CHUNK_;
  const bf16* Gc = G + (long)(b * NCHUNK_ + c) * CHUNK_ * CHUNK_;

  __shared__ bf16 sXcT[HEADDIM_ * LDX];
  __shared__ float sAcum[CHUNK_];
  sAcum[threadIdx.x] = Acum[((long)(b * NHEADS_ + h) * NCHUNK_ + c) * CHUNK_ + threadIdx.x];
  {
    int s = threadIdx.x;
    const bf16* src = xc + (long)(mbase + s) * DINNER + h * HEADDIM_;
#pragma unroll
    for (int p = 0; p < HEADDIM_; ++p) sXcT[p * LDX + s] = src[p];
  }
  __syncthreads();

  const int wave = threadIdx.x >> 5;
  const int lane = threadIdx.x & 31;
  const int q0w = wave * 32;
  const v8f vzero = {0.f, 0.f, 0.f, 0.f, 0.f, 0.f, 0.f, 0.f};
  v8f acc[2][4];
#pragma unroll
  for (int i = 0; i < 2; ++i)
#pragma unroll
    for (int j = 0; j < 4; ++j) acc[i][j] = vzero;

  for (int k0 = 0; k0 <= q0w; k0 += 32) {  // tril: skip fully-masked K blocks
    v16bf bfr[4];
#pragma unroll
    for (int j = 0; j < 4; ++j) bfr[j] = load_b_frag(sXcT + (j * 16) * LDX + k0, LDX);
#pragma unroll
    for (int i = 0; i < 2; ++i) {
      int q = q0w + i * 16 + (lane & 15);
      v16bf af = make_gfrag(Gc + (long)q * CHUNK_, q, k0, sAcum[q], sAcum);
#pragma unroll
      for (int j = 0; j < 4; ++j) acc[i][j] = wmma_bf16(af, bfr[j], acc[i][j]);
    }
  }
#pragma unroll
  for (int i = 0; i < 2; ++i)
#pragma unroll
    for (int j = 0; j < 4; ++j)
#pragma unroll
      for (int r = 0; r < 8; ++r) {
        int q = q0w + i * 16 + ((lane >> 4) << 3) + r;
        int p = j * 16 + (lane & 15);
        y_acc[(long)(mbase + q) * DINNER + h * HEADDIM_ + p] = acc[i][j][r];
      }
}

// ---------------------------------------------------------------- chunk states per (b,c,h)
// states[p,n] = sum_s xc[s,p] * exp(Acum_last - Acum_s) * B[s,n]  (64x128, K=256)
#define LDB2 40

__global__ __launch_bounds__(256) void states_kernel(const bf16* __restrict__ xc,
                                                     const bf16* __restrict__ Bm,
                                                     const float* __restrict__ Acum,
                                                     float* __restrict__ states) {
  const int blk = blockIdx.x;
  const int h = blk & (NHEADS_ - 1);
  const int c = (blk >> 6) & (NCHUNK_ - 1);
  const int b = blk >> 9;
  const int mbase = b * SEQLEN_ + c * CHUNK_;

  __shared__ bf16 sXcT[HEADDIM_ * LDX];   // [p][s]
  __shared__ bf16 sBt[DSTATE * LDB2];     // [n][s_local], decay-scaled
  __shared__ float sAcum[CHUNK_];
  sAcum[threadIdx.x] = Acum[((long)(b * NHEADS_ + h) * NCHUNK_ + c) * CHUNK_ + threadIdx.x];
  {
    int s = threadIdx.x;
    const bf16* src = xc + (long)(mbase + s) * DINNER + h * HEADDIM_;
#pragma unroll
    for (int p = 0; p < HEADDIM_; ++p) sXcT[p * LDX + s] = src[p];
  }
  __syncthreads();
  const float aclast = sAcum[CHUNK_ - 1];

  const int wave = threadIdx.x >> 5;
  const int lane = threadIdx.x & 31;
  const int wm = wave >> 2;  // 0..1 (rows p)
  const int wn = wave & 3;   // 0..3 (cols n)
  const v8f vzero = {0.f, 0.f, 0.f, 0.f, 0.f, 0.f, 0.f, 0.f};
  v8f acc[2][2];
#pragma unroll
  for (int i = 0; i < 2; ++i)
#pragma unroll
    for (int j = 0; j < 2; ++j) acc[i][j] = vzero;

  const int sl = threadIdx.x >> 3;   // 0..31 : s within K tile
  const int sub = threadIdx.x & 7;   // 0..7  : 16-wide n chunk
  for (int k0 = 0; k0 < CHUNK_; k0 += 32) {
    {
      int s = k0 + sl;
      float dec = __expf(aclast - sAcum[s]);
      const bf16* src = Bm + (long)(mbase + s) * DSTATE + sub * 16;
#pragma unroll
      for (int i = 0; i < 16; ++i)
        sBt[(sub * 16 + i) * LDB2 + sl] = f2bf(bf2f(src[i]) * dec);
    }
    __syncthreads();
    v16bf af[2], bfr[2];
#pragma unroll
    for (int i = 0; i < 2; ++i)
      af[i] = load_a_frag(sXcT + (wm * 32 + i * 16) * LDX + k0, LDX);
#pragma unroll
    for (int j = 0; j < 2; ++j)
      bfr[j] = load_b_frag(sBt + (wn * 32 + j * 16) * LDB2, LDB2);
#pragma unroll
    for (int i = 0; i < 2; ++i)
#pragma unroll
      for (int j = 0; j < 2; ++j) acc[i][j] = wmma_bf16(af[i], bfr[j], acc[i][j]);
    __syncthreads();
  }
  float* outp = states + ((long)((b * NCHUNK_ + c) * NHEADS_) + h) * HEADDIM_ * DSTATE;
#pragma unroll
  for (int i = 0; i < 2; ++i)
#pragma unroll
    for (int j = 0; j < 2; ++j)
#pragma unroll
      for (int r = 0; r < 8; ++r) {
        int p = wm * 32 + i * 16 + ((lane >> 4) << 3) + r;
        int n = wn * 32 + j * 16 + (lane & 15);
        outp[(long)p * DSTATE + n] = acc[i][j][r];
      }
}

// ---------------------------------------------------------------- inter-chunk scan
__global__ __launch_bounds__(256) void scan_kernel(const float* __restrict__ states,
                                                   const float* __restrict__ Acum,
                                                   bf16* __restrict__ prev) {
  const int bh = blockIdx.x;
  const int b = bh >> 6, h = bh & (NHEADS_ - 1);
  float dec[NCHUNK_];
#pragma unroll
  for (int c = 0; c < NCHUNK_; ++c)
    dec[c] = __expf(Acum[((long)(b * NHEADS_ + h) * NCHUNK_ + c) * CHUNK_ + CHUNK_ - 1]);
  for (int pn = threadIdx.x; pn < HEADDIM_ * DSTATE; pn += 256) {
    float S = 0.f;
#pragma unroll
    for (int c = 0; c < NCHUNK_; ++c) {
      long idx = ((long)((b * NCHUNK_ + c) * NHEADS_) + h) * HEADDIM_ * DSTATE + pn;
      prev[idx] = f2bf(S);
      S = dec[c] * S + states[idx];
    }
  }
}

// ---------------------------------------------------------------- Y_off + D*x per (b,c,h)
// Y_off[q,p] = sum_n C[q,n]*exp(Acum_q) * prev[p,n]  (256x64, K=128); adds into y_acc.
__global__ __launch_bounds__(256) void yoff_kernel(const bf16* __restrict__ Cm,
                                                   const bf16* __restrict__ prev,
                                                   const float* __restrict__ Acum,
                                                   const bf16* __restrict__ xc,
                                                   const float* __restrict__ dtv,
                                                   const float* __restrict__ Dp,
                                                   float* __restrict__ y_acc) {
  const int blk = blockIdx.x;
  const int h = blk & (NHEADS_ - 1);
  const int c = (blk >> 6) & (NCHUNK_ - 1);
  const int b = blk >> 9;
  const int mbase = b * SEQLEN_ + c * CHUNK_;

  __shared__ float sAcum[CHUNK_];
  sAcum[threadIdx.x] = Acum[((long)(b * NHEADS_ + h) * NCHUNK_ + c) * CHUNK_ + threadIdx.x];
  __syncthreads();

  const bf16* pv = prev + ((long)((b * NCHUNK_ + c) * NHEADS_) + h) * HEADDIM_ * DSTATE;
  const bf16* Cc = Cm + (long)mbase * DSTATE;
  const int wave = threadIdx.x >> 5;
  const int lane = threadIdx.x & 31;
  const int q0w = wave * 32;
  const v8f vzero = {0.f, 0.f, 0.f, 0.f, 0.f, 0.f, 0.f, 0.f};
  v8f acc[2][4];
#pragma unroll
  for (int i = 0; i < 2; ++i)
#pragma unroll
    for (int j = 0; j < 4; ++j) acc[i][j] = vzero;

  for (int k0 = 0; k0 < DSTATE; k0 += 32) {
    v16bf bfr[4];
#pragma unroll
    for (int j = 0; j < 4; ++j) bfr[j] = load_b_frag(pv + (long)(j * 16) * DSTATE + k0, DSTATE);
#pragma unroll
    for (int i = 0; i < 2; ++i) {
      int q = q0w + i * 16 + (lane & 15);
      float eq = __expf(sAcum[q]);
      v16bf af = make_scaled_afrag(Cc + (long)q * DSTATE, k0, eq);
#pragma unroll
      for (int j = 0; j < 4; ++j) acc[i][j] = wmma_bf16(af, bfr[j], acc[i][j]);
    }
  }
  const float Dv = Dp[h];
#pragma unroll
  for (int i = 0; i < 2; ++i)
#pragma unroll
    for (int j = 0; j < 4; ++j)
#pragma unroll
      for (int r = 0; r < 8; ++r) {
        int q = q0w + i * 16 + ((lane >> 4) << 3) + r;
        int p = j * 16 + (lane & 15);
        long mi = mbase + q;
        float xval = bf2f(xc[mi * DINNER + h * HEADDIM_ + p]) / dtv[mi * NHEADS_ + h];
        long oi = mi * DINNER + h * HEADDIM_ + p;
        y_acc[oi] += acc[i][j][r] + Dv * xval;
      }
}

// ---------------------------------------------------------------- gate + RMSNorm
__global__ __launch_bounds__(256) void gate_norm_kernel(const float* __restrict__ y_acc,
                                                        const bf16* __restrict__ z,
                                                        const float* __restrict__ norm_w,
                                                        bf16* __restrict__ y_out) {
  const int m = blockIdx.x;
  const int tid = threadIdx.x;
  float v[DINNER / 256];
  float ss = 0.f;
#pragma unroll
  for (int t = 0; t < DINNER / 256; ++t) {
    int i = tid + t * 256;
    float zv = bf2f(z[(long)m * DINNER + i]);
    float gate = zv / (1.f + __expf(-zv));
    float val = y_acc[(long)m * DINNER + i] * gate;
    v[t] = val;
    ss += val * val;
  }
#pragma unroll
  for (int o = 16; o > 0; o >>= 1) ss += __shfl_xor(ss, o, 32);
  __shared__ float sred[8];
  __shared__ float sscale;
  if ((tid & 31) == 0) sred[tid >> 5] = ss;
  __syncthreads();
  if (tid == 0) {
    float tot = 0.f;
    for (int w = 0; w < 8; ++w) tot += sred[w];
    sscale = rsqrtf(tot / (float)DINNER + 1e-5f);
  }
  __syncthreads();
  const float scale = sscale;
#pragma unroll
  for (int t = 0; t < DINNER / 256; ++t) {
    int i = tid + t * 256;
    y_out[(long)m * DINNER + i] = f2bf(v[t] * scale * norm_w[i]);
  }
}

// ---------------------------------------------------------------- launch
extern "C" void kernel_launch(void* const* d_in, const int* in_sizes, int n_in,
                              void* d_out, int out_size, void* d_ws, size_t ws_size,
                              hipStream_t stream) {
  (void)in_sizes; (void)n_in; (void)out_size; (void)ws_size;
  const float* u          = (const float*)d_in[0];
  const float* in_proj_w  = (const float*)d_in[1];
  const float* conv_w     = (const float*)d_in[2];
  const float* conv_b     = (const float*)d_in[3];
  const float* dt_bias    = (const float*)d_in[4];
  const float* A_log      = (const float*)d_in[5];
  const float* Dparam     = (const float*)d_in[6];
  const float* norm_w     = (const float*)d_in[7];
  const float* out_proj_w = (const float*)d_in[8];
  float* out = (float*)d_out;

  char* ws = (char*)d_ws;
  size_t off = 0;
  auto take = [&](size_t bytes) -> char* {
    char* p = ws + off;
    off = (off + bytes + 255) & ~(size_t)255;
    return p;
  };
  const long n_u = (long)BL_ * DMODEL;
  const long n_win = (long)DPROJ * DMODEL;
  const long n_wout = (long)DMODEL * DINNER;

  bf16*  u_bf     = (bf16*)take(n_u * 2);
  bf16*  w_in_bf  = (bf16*)take(n_win * 2);
  bf16*  w_out_bf = (bf16*)take(n_wout * 2);
  bf16*  z_bf     = (bf16*)take((long)BL_ * DINNER * 2);
  float* xbc      = (float*)take((long)BL_ * CONVDIM * 4);
  float* dt_raw   = (float*)take((long)BL_ * NHEADS_ * 4);
  float* dtv_p    = (float*)take((long)BL_ * NHEADS_ * 4);
  float* acum_p   = (float*)take((long)BATCH_ * NHEADS_ * NCHUNK_ * CHUNK_ * 4);
  bf16*  xc_p     = (bf16*)take((long)BL_ * DINNER * 2);
  bf16*  Bm_p     = (bf16*)take((long)BL_ * DSTATE * 2);
  bf16*  Cm_p     = (bf16*)take((long)BL_ * DSTATE * 2);
  bf16*  G_p      = (bf16*)take((long)BATCH_ * NCHUNK_ * CHUNK_ * CHUNK_ * 2);
  float* y_acc_p  = (float*)take((long)BL_ * DINNER * 4);
  float* states_p = (float*)take((long)BATCH_ * NCHUNK_ * NHEADS_ * HEADDIM_ * DSTATE * 4);
  bf16*  prev_p   = (bf16*)take((long)BATCH_ * NCHUNK_ * NHEADS_ * HEADDIM_ * DSTATE * 2);
  bf16*  y_bf     = (bf16*)take((long)BL_ * DINNER * 2);

  // 1) bf16 conversions
  f32_to_bf16_kernel<<<(unsigned)((n_u + 255) / 256), 256, 0, stream>>>(u, u_bf, n_u);
  f32_to_bf16_kernel<<<(unsigned)((n_win + 255) / 256), 256, 0, stream>>>(in_proj_w, w_in_bf, n_win);
  f32_to_bf16_kernel<<<(unsigned)((n_wout + 255) / 256), 256, 0, stream>>>(out_proj_w, w_out_bf, n_wout);

  // 2) in_proj GEMM with split epilogue (z | xBC | dt); N=8512 -> 67 ragged col blocks
  mamba_gemm_nt<<<dim3(BL_ / GBM, (DPROJ + GBN - 1) / GBN, 1), 256, 0, stream>>>(
      u_bf, w_in_bf, BL_, DPROJ, DMODEL, 0, 0, 0, /*mode=*/1,
      nullptr, nullptr, z_bf, xbc, dt_raw);

  // 3) softplus + per-chunk cumsum
  dtv_kernel<<<(BL_ * NHEADS_ + 255) / 256, 256, 0, stream>>>(dt_raw, dt_bias, dtv_p);
  acum_kernel<<<BATCH_ * NCHUNK_ * NHEADS_, CHUNK_, 0, stream>>>(dtv_p, A_log, acum_p);

  // 4) depthwise conv + silu + split into xc (x*dt), B, C (bf16)
  convprep_kernel<<<(unsigned)(((long)BL_ * CONVDIM + 255) / 256), 256, 0, stream>>>(
      xbc, conv_w, conv_b, dtv_p, xc_p, Bm_p, Cm_p);

  // 5) G = C @ B^T per chunk (batched over 16 chunks)
  mamba_gemm_nt<<<dim3(CHUNK_ / GBM, CHUNK_ / GBN, BATCH_ * NCHUNK_), 256, 0, stream>>>(
      Cm_p, Bm_p, CHUNK_, CHUNK_, DSTATE,
      (long)CHUNK_ * DSTATE, (long)CHUNK_ * DSTATE, (long)CHUNK_ * CHUNK_, /*mode=*/2,
      nullptr, G_p, nullptr, nullptr, nullptr);

  // 6) intra-chunk output, chunk states, inter-chunk scan, cross-chunk output
  ydiag_kernel<<<BATCH_ * NCHUNK_ * NHEADS_, 256, 0, stream>>>(G_p, xc_p, acum_p, y_acc_p);
  states_kernel<<<BATCH_ * NCHUNK_ * NHEADS_, 256, 0, stream>>>(xc_p, Bm_p, acum_p, states_p);
  scan_kernel<<<BATCH_ * NHEADS_, 256, 0, stream>>>(states_p, acum_p, prev_p);
  yoff_kernel<<<BATCH_ * NCHUNK_ * NHEADS_, 256, 0, stream>>>(
      Cm_p, prev_p, acum_p, xc_p, dtv_p, Dparam, y_acc_p);

  // 7) silu(z) gate + RMSNorm -> bf16
  gate_norm_kernel<<<BL_, 256, 0, stream>>>(y_acc_p, z_bf, norm_w, y_bf);

  // 8) out_proj GEMM -> f32 output
  mamba_gemm_nt<<<dim3(BL_ / GBM, DMODEL / GBN, 1), 256, 0, stream>>>(
      y_bf, w_out_bf, BL_, DMODEL, DINNER, 0, 0, 0, /*mode=*/0,
      out, nullptr, nullptr, nullptr, nullptr);
}
